// Attention_43937515438607
// MI455X (gfx1250) — matrix operand order repaired and checked by reading, not compiled, not run on hardware
//
#include <hip/hip_runtime.h>

// ---------------------------------------------------------------------------
// Problem constants (match reference)
// ---------------------------------------------------------------------------
#define B_   16
#define T_   1024
#define C_   512
#define NH_  8
#define D_   64
#define M_   (B_ * T_)      // 16384 rows of x
#define N3C_ (3 * C_)       // 1536 qkv columns

constexpr float kInvTP = 0.08838834764831845f; // 1/sqrt(2*D)

typedef _Float16 half16  __attribute__((ext_vector_type(16)));
typedef _Float16 half8   __attribute__((ext_vector_type(8)));
typedef float    floatx8 __attribute__((ext_vector_type(8)));

union Frag16 {           // one WMMA f16 A- or B-operand (8 VGPRs)
    half16 v;
    half8  h[2];
};

__device__ __forceinline__ floatx8 wmma_f16(const Frag16& a, const Frag16& b,
                                            floatx8 c) {
    // D = A(16x32) * B(32x16) + C, f32 accumulate
    return __builtin_amdgcn_wmma_f32_16x16x32_f16(
        /*neg_a=*/false, a.v, /*neg_b=*/false, b.v,
        /*c_mod=*/(short)0, c, /*reuse_a=*/false, /*reuse_b=*/false);
}

// ---------------------------------------------------------------------------
// Async global->LDS copy (CDNA5 GLOBAL_LOAD_ASYNC_TO_LDS_B128, ASYNCcnt),
// with a synchronous fallback if the builtins are not exposed.
// Builtin signature (from hipcc diagnostic): params are pointers to
// 16-byte int vectors, global (AS1) source and LDS (AS3) destination.
// ---------------------------------------------------------------------------
#if defined(__has_builtin)
#if __has_builtin(__builtin_amdgcn_global_load_async_to_lds_b128) && \
    __has_builtin(__builtin_amdgcn_s_wait_asynccnt)
#define HAVE_ASYNC_LDS 1
#endif
#endif

typedef int v4i __attribute__((vector_size(4 * sizeof(int))));
typedef __attribute__((address_space(1))) v4i as1_v4i;   // global int4
typedef __attribute__((address_space(3))) v4i as3_v4i;   // LDS int4

__device__ __forceinline__ void async_cp16(const _Float16* g, _Float16* l) {
#ifdef HAVE_ASYNC_LDS
    __builtin_amdgcn_global_load_async_to_lds_b128(
        (as1_v4i*)g, (as3_v4i*)l, /*offset=*/0, /*cpol=*/0);
#else
    *(half8*)l = *(const half8*)g;
#endif
}

// ---------------------------------------------------------------------------
// Kernel 0: f32 -> f16 conversion (x and W_qkv)
// ---------------------------------------------------------------------------
__global__ __launch_bounds__(256) void cvt_f32_to_f16(const float* __restrict__ src,
                                                      _Float16* __restrict__ dst,
                                                      int n) {
    int i = blockIdx.x * 256 + threadIdx.x;
    if (i < n) dst[i] = (_Float16)src[i];
}

// ---------------------------------------------------------------------------
// Kernel 1: qkv = x @ W^T + b, fused RoPE, scatter into q/k ([H][B][T][D]) and
// vT ([H][B][D][T]) in f16. One wave computes a 16(M) x 64(N) tile: per K-step
// one A-fragment is reused by 4 independent WMMAs into 4 accumulators.
// ---------------------------------------------------------------------------
__global__ __launch_bounds__(256) void qkv_gemm_rope(
    const _Float16* __restrict__ xb,   // [M_][C_] row-major f16
    const _Float16* __restrict__ wb,   // [N3C_][C_] row-major f16
    const float* __restrict__ bias,    // [N3C_]
    const float* __restrict__ pe0,     // [T_][D_]
    const float* __restrict__ pe1,     // [T_][D_]
    const float* __restrict__ pe2,     // [T_][D_]
    _Float16* __restrict__ qh,         // [NH_][B_][T_][D_]
    _Float16* __restrict__ kh,         // [NH_][B_][T_][D_]
    _Float16* __restrict__ vT)         // [NH_][B_][D_][T_]
{
    const int lane = threadIdx.x & 31;
    const int wid  = threadIdx.x >> 5;
    const int tile = blockIdx.x * 8 + wid;       // (M/16)*(N/64) = 24576 tiles
    const int NTN  = N3C_ / 64;                  // 24
    const int tn = tile % NTN;
    const int tm = tile / NTN;
    const int m0 = tm * 16, n0 = tn * 64;
    const int hi = lane >> 4;                    // half-group 0/1
    const int li = lane & 15;

    const _Float16* arow = xb + (size_t)(m0 + li) * C_ + hi * 8;
    const _Float16* brow0 = wb + (size_t)(n0 + li) * C_ + hi * 8;
    const _Float16* brow1 = brow0 + (size_t)16 * C_;
    const _Float16* brow2 = brow0 + (size_t)32 * C_;
    const _Float16* brow3 = brow0 + (size_t)48 * C_;

    floatx8 acc[4] = {};
    #pragma unroll 2
    for (int kc = 0; kc < C_; kc += 32) {
        Frag16 a, b;
        a.h[0] = *(const half8*)(arow + kc);       // K = kc + hi*8 + 0..7
        a.h[1] = *(const half8*)(arow + kc + 16);  // K = kc + 16 + hi*8 + 0..7
        b.h[0] = *(const half8*)(brow0 + kc);
        b.h[1] = *(const half8*)(brow0 + kc + 16);
        acc[0] = wmma_f16(a, b, acc[0]);
        b.h[0] = *(const half8*)(brow1 + kc);
        b.h[1] = *(const half8*)(brow1 + kc + 16);
        acc[1] = wmma_f16(a, b, acc[1]);
        b.h[0] = *(const half8*)(brow2 + kc);
        b.h[1] = *(const half8*)(brow2 + kc + 16);
        acc[2] = wmma_f16(a, b, acc[2]);
        b.h[0] = *(const half8*)(brow3 + kc);
        b.h[1] = *(const half8*)(brow3 + kc + 16);
        acc[3] = wmma_f16(a, b, acc[3]);
    }

    // ---- epilogue: bias + RoPE + scatter (region/head are wave-uniform) ----
    const int region = n0 >> 9;          // 0=q, 1=k, 2=v  (64 | 512)
    #pragma unroll
    for (int j = 0; j < 4; ++j) {
        const int n  = n0 + j * 16 + li;
        const int c  = n & (C_ - 1);
        const int hd = c >> 6;           // head (uniform: 64-col tile, 64=D)
        const int d  = c & (D_ - 1);
        const float bn = bias[n];
        #pragma unroll
        for (int r = 0; r < 8; ++r) {
            const int m  = m0 + hi * 8 + r;
            const int bb = m / T_;
            const int t  = m % T_;
            float val = acc[j][r] + bn;
            if (region < 2) {
                // rotation pair partner lives in lane^1 (adjacent column)
                float other = __shfl_xor(val, 1, 32);
                float rot   = (d & 1) ? other : -other;   // rotate_half
                const int pidx = t * D_ + d;
                float res = val * pe0[pidx] + rot * pe1[pidx];
                res = (region == 0) ? res * pe2[pidx] : res / pe2[pidx];
                _Float16* dst = (region == 0) ? qh : kh;
                dst[(((size_t)hd * B_ + bb) * T_ + t) * D_ + d] = (_Float16)res;
            } else {
                vT[(((size_t)hd * B_ + bb) * D_ + d) * T_ + t] = (_Float16)val;
            }
        }
    }
}

// ---------------------------------------------------------------------------
// Kernel 2: flash attention. One block (8 waves) owns one (head,batch) and 128
// query rows; wave w handles queries [q0+16w, q0+16w+16). Per 32-key step the
// block cooperatively stages k/v tiles into LDS (async-to-LDS, double
// buffered), then each wave runs 4 score WMMAs + online softmax + 4 alpha*v
// WMMAs. Alpha tiles are re-fragmented C->A through per-wave LDS.
// ---------------------------------------------------------------------------
__global__ __launch_bounds__(256) void flash_attn(
    const _Float16* __restrict__ qh,
    const _Float16* __restrict__ kh,
    const _Float16* __restrict__ vT,
    const unsigned char* __restrict__ mask,  // [B_][T_][T_], nonzero = masked
    float* __restrict__ out)                 // [B_][T_][C_]
{
    __shared__ _Float16 kbuf[2][32][72];     // 32 keys x 64 d (+pad), 16B rows
    __shared__ _Float16 vbuf[2][64][40];     // 64 d x 32 keys (+pad), 16B rows
    __shared__ _Float16 pshare[8][16][40];   // per-wave alpha tile

    const int tid  = threadIdx.x;
    const int lane = tid & 31;
    const int wid  = tid >> 5;
    const int hi   = lane >> 4;
    const int li   = lane & 15;

    const int QB  = T_ / 128;                // 8 query-blocks per (b,h)
    const int hd  = blockIdx.x / (B_ * QB);
    const int rem = blockIdx.x % (B_ * QB);
    const int bb  = rem / QB;
    const int q0  = (rem % QB) * 128 + wid * 16;

    const _Float16* qbase = qh + ((size_t)hd * B_ + bb) * T_ * D_;
    const _Float16* kbase = kh + ((size_t)hd * B_ + bb) * T_ * D_;
    const _Float16* vbase = vT + ((size_t)hd * B_ + bb) * D_ * T_;
    const unsigned char* mbase = mask + (size_t)bb * T_ * T_;

    // ---- cooperative k/v staging: 2 x 16B chunks per thread per stage ----
    const int krow = tid >> 3, kch = (tid & 7) * 8;   // 32 rows x 8 chunks
    const int vrow = tid >> 2, vch = (tid & 3) * 8;   // 64 rows x 4 chunks
    auto stage = [&](int it) {
        const int s0  = it * 32;
        const int buf = it & 1;
        async_cp16(kbase + (size_t)(s0 + krow) * D_ + kch, &kbuf[buf][krow][kch]);
        async_cp16(vbase + (size_t)vrow * T_ + s0 + vch,   &vbuf[buf][vrow][vch]);
    };

    // ---- q A-fragments for d-chunks 0..31 / 32..63 (live whole kernel) ----
    Frag16 aq0, aq1;
    {
        const _Float16* qr = qbase + (size_t)(q0 + li) * D_ + hi * 8;
        aq0.h[0] = *(const half8*)(qr);
        aq0.h[1] = *(const half8*)(qr + 16);
        aq1.h[0] = *(const half8*)(qr + 32);
        aq1.h[1] = *(const half8*)(qr + 48);
    }

    floatx8 acc0 = {}, acc1 = {}, acc2 = {}, acc3 = {};
    float m_r[8], l_r[8];
    #pragma unroll
    for (int r = 0; r < 8; ++r) { m_r[r] = -1e30f; l_r[r] = 0.0f; }

    _Float16 (*P)[40] = pshare[wid];
    const int NIT = T_ / 32;                 // 32 key steps

    stage(0);
    for (int it = 0; it < NIT; ++it) {
        const int s0  = it * 32;
        const int buf = it & 1;
        if (it + 1 < NIT) {
            stage(it + 1);                   // prefetch next tile (other buffer)
#ifdef HAVE_ASYNC_LDS
            __builtin_amdgcn_s_wait_asynccnt(2);  // retire current stage only
#endif
        } else {
#ifdef HAVE_ASYNC_LDS
            __builtin_amdgcn_s_wait_asynccnt(0);
#endif
        }
        __syncthreads();                     // staged tile visible to block

        // ---- scores: S(16q x 32k), two 16-key accumulators ----
        floatx8 s0acc = {}, s1acc = {};
        {
            Frag16 bk;
            bk.h[0] = *(const half8*)(&kbuf[buf][li][hi * 8]);
            bk.h[1] = *(const half8*)(&kbuf[buf][li][hi * 8 + 16]);
            s0acc = wmma_f16(aq0, bk, s0acc);
            bk.h[0] = *(const half8*)(&kbuf[buf][li][32 + hi * 8]);
            bk.h[1] = *(const half8*)(&kbuf[buf][li][32 + hi * 8 + 16]);
            s0acc = wmma_f16(aq1, bk, s0acc);
            bk.h[0] = *(const half8*)(&kbuf[buf][16 + li][hi * 8]);
            bk.h[1] = *(const half8*)(&kbuf[buf][16 + li][hi * 8 + 16]);
            s1acc = wmma_f16(aq0, bk, s1acc);
            bk.h[0] = *(const half8*)(&kbuf[buf][16 + li][32 + hi * 8]);
            bk.h[1] = *(const half8*)(&kbuf[buf][16 + li][32 + hi * 8 + 16]);
            s1acc = wmma_f16(aq1, bk, s1acc);
        }

        // ---- scale + mask (C layout: elem r -> row hi*8+r, col li) ----
        const int col0 = s0 + li, col1 = s0 + 16 + li;
        float sv0[8], sv1[8], mnew[8];
        #pragma unroll
        for (int r = 0; r < 8; ++r) {
            const int t = q0 + hi * 8 + r;
            const unsigned char* mr = mbase + (size_t)t * T_;
            float x0 = s0acc[r] * kInvTP;
            float x1 = s1acc[r] * kInvTP;
            if (mr[col0]) x0 = -3.0e38f;
            if (mr[col1]) x1 = -3.0e38f;
            sv0[r] = x0; sv1[r] = x1;
        }
        // ---- online softmax: row max over the 16-lane half-group ----
        #pragma unroll
        for (int r = 0; r < 8; ++r) {
            float mx = fmaxf(sv0[r], sv1[r]);
            mx = fmaxf(mx, __shfl_xor(mx, 1, 32));
            mx = fmaxf(mx, __shfl_xor(mx, 2, 32));
            mx = fmaxf(mx, __shfl_xor(mx, 4, 32));
            mx = fmaxf(mx, __shfl_xor(mx, 8, 32));
            mnew[r] = fmaxf(m_r[r], mx);
        }
        #pragma unroll
        for (int r = 0; r < 8; ++r) {
            float p0   = __expf(sv0[r] - mnew[r]);
            float p1   = __expf(sv1[r] - mnew[r]);
            float corr = __expf(m_r[r] - mnew[r]);
            float rs = p0 + p1;
            rs += __shfl_xor(rs, 1, 32);
            rs += __shfl_xor(rs, 2, 32);
            rs += __shfl_xor(rs, 4, 32);
            rs += __shfl_xor(rs, 8, 32);
            l_r[r] = l_r[r] * corr + rs;
            m_r[r] = mnew[r];
            acc0[r] *= corr; acc1[r] *= corr; acc2[r] *= corr; acc3[r] *= corr;
            const int row = hi * 8 + r;
            P[row][li]      = (_Float16)p0;   // alpha tile, C layout -> LDS
            P[row][16 + li] = (_Float16)p1;
        }

        // ---- re-fragment alpha into A layout (same-wave LDS round trip) ----
        Frag16 aP;
        aP.h[0] = *(const half8*)(&P[li][hi * 8]);
        aP.h[1] = *(const half8*)(&P[li][16 + hi * 8]);

        // ---- out += alpha(16x32) * v(32keys x 64d), 4 d-tiles ----
        Frag16 bv;
        bv.h[0] = *(const half8*)(&vbuf[buf][li][hi * 8]);
        bv.h[1] = *(const half8*)(&vbuf[buf][li][hi * 8 + 16]);
        acc0 = wmma_f16(aP, bv, acc0);
        bv.h[0] = *(const half8*)(&vbuf[buf][16 + li][hi * 8]);
        bv.h[1] = *(const half8*)(&vbuf[buf][16 + li][hi * 8 + 16]);
        acc1 = wmma_f16(aP, bv, acc1);
        bv.h[0] = *(const half8*)(&vbuf[buf][32 + li][hi * 8]);
        bv.h[1] = *(const half8*)(&vbuf[buf][32 + li][hi * 8 + 16]);
        acc2 = wmma_f16(aP, bv, acc2);
        bv.h[0] = *(const half8*)(&vbuf[buf][48 + li][hi * 8]);
        bv.h[1] = *(const half8*)(&vbuf[buf][48 + li][hi * 8 + 16]);
        acc3 = wmma_f16(aP, bv, acc3);

        __syncthreads();                     // tile consumed; buffer reusable
    }

    // ---- finalize: out[b][t][h*64+d] = acc/l ----
    #pragma unroll
    for (int r = 0; r < 8; ++r) {
        const int t = q0 + hi * 8 + r;
        const float inv_l = 1.0f / l_r[r];
        float* orow = out + ((size_t)bb * T_ + t) * C_ + hd * D_;
        orow[li]      = acc0[r] * inv_l;
        orow[16 + li] = acc1[r] * inv_l;
        orow[32 + li] = acc2[r] * inv_l;
        orow[48 + li] = acc3[r] * inv_l;
    }
}

// ---------------------------------------------------------------------------
// Host-side launcher
// ---------------------------------------------------------------------------
extern "C" void kernel_launch(void* const* d_in, const int* in_sizes, int n_in,
                              void* d_out, int out_size, void* d_ws, size_t ws_size,
                              hipStream_t stream) {
    const float*         x     = (const float*)d_in[0];
    const float*         pe0   = (const float*)d_in[1];
    const float*         pe1   = (const float*)d_in[2];
    const float*         pe2   = (const float*)d_in[3];
    const unsigned char* mask  = (const unsigned char*)d_in[4];  // bool array
    const float*         W_qkv = (const float*)d_in[5];
    const float*         b_qkv = (const float*)d_in[6];
    float*               out   = (float*)d_out;

    // workspace layout (f16 intermediates; total ~68.7 MB)
    char* ws = (char*)d_ws;
    size_t off = 0;
    _Float16* xb = (_Float16*)(ws + off); off += (size_t)M_ * C_ * 2;      // 16 MB
    _Float16* wb = (_Float16*)(ws + off); off += (size_t)N3C_ * C_ * 2;    // 1.5 MB
    _Float16* qh = (_Float16*)(ws + off); off += (size_t)NH_*B_*T_*D_ * 2; // 16 MB
    _Float16* kh = (_Float16*)(ws + off); off += (size_t)NH_*B_*T_*D_ * 2; // 16 MB
    _Float16* vT = (_Float16*)(ws + off); off += (size_t)NH_*B_*T_*D_ * 2; // 16 MB
    (void)ws_size; (void)in_sizes; (void)n_in; (void)out_size;

    // 1) convert x, W to f16
    {
        int nx = M_ * C_;
        int nw = N3C_ * C_;
        cvt_f32_to_f16<<<(nx + 255) / 256, 256, 0, stream>>>(x, xb, nx);
        cvt_f32_to_f16<<<(nw + 255) / 256, 256, 0, stream>>>(W_qkv, wb, nw);
    }
    // 2) qkv GEMM + RoPE + scatter: 24576 wave-tiles (16x64), 8 waves/block
    {
        int tiles  = (M_ / 16) * (N3C_ / 64);   // 1024 * 24
        int blocks = tiles / 8;                  // 3072
        qkv_gemm_rope<<<blocks, 256, 0, stream>>>(xb, wb, b_qkv, pe0, pe1, pe2,
                                                  qh, kh, vT);
    }
    // 3) flash attention: one block per (head, batch, 128-query slab)
    {
        int blocks = NH_ * B_ * (T_ / 128);     // 1024
        flash_attn<<<blocks, 256, 0, stream>>>(qh, kh, vT, mask, out);
    }
}
